// RingDilatedAttentionHybridOptimized_42614665511001
// MI455X (gfx1250) — matrix operand order repaired and checked by reading, not compiled, not run on hardware
//
#include <hip/hip_runtime.h>
#include <math.h>
#include <stdint.h>

typedef __attribute__((ext_vector_type(16))) __bf16 v16bf;
typedef __attribute__((ext_vector_type(8)))  __bf16 v8bf;
typedef __attribute__((ext_vector_type(8)))  float  v8f;

#define NHEAD 12
#define DHEAD 64
#define ROWB  (NHEAD * DHEAD * 4)       // 3072 bytes per sequence position
#define KVSTR 68                        // f32 elems per LDS K/V row (64 + 4 pad)
#define PSTR  40                        // bf16 elems per LDS P row (32 + 8 pad)
#define WLDSB (2 * 32 * KVSTR * 4 + 32 * PSTR * 2)   // 17408 + 2560 = 19968 B per wave

// cross-lane reductions within each 16-lane half (matches WMMA C/D row layout)
__device__ __forceinline__ float rmax16(float x) {
  x = fmaxf(x, __int_as_float(__builtin_amdgcn_ds_swizzle(__float_as_int(x), 0x041f)));
  x = fmaxf(x, __int_as_float(__builtin_amdgcn_ds_swizzle(__float_as_int(x), 0x081f)));
  x = fmaxf(x, __int_as_float(__builtin_amdgcn_ds_swizzle(__float_as_int(x), 0x101f)));
  x = fmaxf(x, __int_as_float(__builtin_amdgcn_ds_swizzle(__float_as_int(x), 0x201f)));
  return x;
}
__device__ __forceinline__ float rsum16(float x) {
  x += __int_as_float(__builtin_amdgcn_ds_swizzle(__float_as_int(x), 0x041f));
  x += __int_as_float(__builtin_amdgcn_ds_swizzle(__float_as_int(x), 0x081f));
  x += __int_as_float(__builtin_amdgcn_ds_swizzle(__float_as_int(x), 0x101f));
  x += __int_as_float(__builtin_amdgcn_ds_swizzle(__float_as_int(x), 0x201f));
  return x;
}

__device__ __forceinline__ v16bf cat16(v8bf lo, v8bf hi) {
  return __builtin_shufflevector(lo, hi, 0,1,2,3,4,5,6,7,8,9,10,11,12,13,14,15);
}
__device__ __forceinline__ v8f wmma_bf16(v16bf a, v16bf b, v8f c) {
  return __builtin_amdgcn_wmma_f32_16x16x32_bf16(false, a, false, b, (short)0, c, false, false);
}
// raw v_exp_f32: args are always <= 0 here; sub-denormal results are ~0 probs anyway
__device__ __forceinline__ float fast_exp2(float x) { return __builtin_amdgcn_exp2f(x); }

// convert two contiguous 8-float chunks into a v16bf A/B fragment
__device__ __forceinline__ v16bf cvt8x2(const float* plo, const float* phi, float s) {
  const float4* l = (const float4*)plo;
  const float4* h = (const float4*)phi;
  float4 a = l[0], b = l[1], c = h[0], d = h[1];
  v16bf r;
  r[0]=(__bf16)(a.x*s);  r[1]=(__bf16)(a.y*s);  r[2]=(__bf16)(a.z*s);  r[3]=(__bf16)(a.w*s);
  r[4]=(__bf16)(b.x*s);  r[5]=(__bf16)(b.y*s);  r[6]=(__bf16)(b.z*s);  r[7]=(__bf16)(b.w*s);
  r[8]=(__bf16)(c.x*s);  r[9]=(__bf16)(c.y*s);  r[10]=(__bf16)(c.z*s); r[11]=(__bf16)(c.w*s);
  r[12]=(__bf16)(d.x*s); r[13]=(__bf16)(d.y*s); r[14]=(__bf16)(d.z*s); r[15]=(__bf16)(d.w*s);
  return r;
}

__global__ void dilated_zero_kernel(float* __restrict__ o, int n) {
  int i = blockIdx.x * blockDim.x + threadIdx.x;
  if (i < n) o[i] = 0.0f;
}

__global__ __launch_bounds__(64)
void dilated_attn_wmma_kernel(const float* __restrict__ Q, const float* __restrict__ K,
                              const float* __restrict__ V, float* __restrict__ O) {
  __shared__ __align__(16) unsigned char ldsraw[2 * WLDSB];
  const int wid  = threadIdx.x >> 5;
  const int lane = threadIdx.x & 31;
  unsigned char* slab = ldsraw + wid * WLDSB;
  float*  Kf = (float*)slab;                  // 32 x KVSTR f32
  float*  Vf = Kf + 32 * KVSTR;               // 32 x KVSTR f32
  __bf16* Pb = (__bf16*)(Vf + 32 * KVSTR);    // 32 x PSTR bf16
  const uint32_t kbase = (uint32_t)(uintptr_t)Kf;   // low 32 bits of generic addr == LDS offset
  const uint32_t vbase = (uint32_t)(uintptr_t)Vf;

  // tile (32 q-rows each) -> (head, dilation r, offset, segment start, dilated len, q base)
  const int tile = blockIdx.x * 2 + wid;      // 0..895 exactly
  int head, r, off, seg0, len, q0;
  if (tile < 512)      { head = tile >> 7;                 r = 1; off = 0; seg0 = 0;                len = 4096; q0 = (tile & 127) << 5; }
  else if (tile < 768) { int u = tile - 512; head = 4 + (u >> 6); int u2 = u & 63;
                         r = 2; off = 1; seg0 = (u2 >> 5) << 11;  len = 1024; q0 = (u2 & 31) << 5; }
  else                 { int u = tile - 768; head = 8 + (u >> 5);
                         r = 4; off = 2; seg0 = 0;                len = 1024; q0 = (u & 31) << 5; }

  const int m  = lane & 15;   // fragment row/column index
  const int hh = lane >> 4;   // wave half

  // ---- Q A-fragments for both 16-row q tiles; fold in 1/sqrt(d) * log2(e)
  const float qscale = 0.125f * 1.4426950408889634f;
  v16bf qa[2][2];
  #pragma unroll
  for (int qt = 0; qt < 2; qt++) {
    const float* qrow = Q + ((size_t)(seg0 + off + (q0 + qt * 16 + m) * r) * NHEAD + head) * DHEAD;
    #pragma unroll
    for (int c = 0; c < 2; c++)
      qa[qt][c] = cvt8x2(qrow + c * 32 + hh * 8, qrow + c * 32 + 16 + hh * 8, qscale);
  }

  v8f of[2][4];
  float mst[2][8], lst[2][8];
  #pragma unroll
  for (int qt = 0; qt < 2; qt++)
    #pragma unroll
    for (int i = 0; i < 8; i++) {
      mst[qt][i] = -1e30f; lst[qt][i] = 0.0f;
      if (i < 4) of[qt][i] = v8f{0};
    }

  for (int kv0 = 0; kv0 < len; kv0 += 32) {
    // ensure all outstanding DS reads of the K/V slabs are done before the async
    // engine overwrites them (async LDS writes are not ordered with the DS queue)
    asm volatile("s_wait_dscnt 0" ::: "memory");

    // ---- async-stage 32 K rows + 32 V rows (raw f32, one dilated row per lane)
    {
      const uint32_t grow = (uint32_t)((seg0 + off + (kv0 + lane) * r) * ROWB + head * (DHEAD * 4));
      const uint32_t kld = kbase + (uint32_t)(lane * (KVSTR * 4));
      const uint32_t vld = vbase + (uint32_t)(lane * (KVSTR * 4));
      #pragma unroll
      for (int j = 0; j < 16; j++) {
        asm volatile("global_load_async_to_lds_b128 %0, %1, %2"
                     :: "v"(kld + (uint32_t)(j * 16)), "v"(grow + (uint32_t)(j * 16)), "s"(K)
                     : "memory");
        asm volatile("global_load_async_to_lds_b128 %0, %1, %2"
                     :: "v"(vld + (uint32_t)(j * 16)), "v"(grow + (uint32_t)(j * 16)), "s"(V)
                     : "memory");
      }
      // prefetch next chunk's rows into cache while this chunk computes
      if (kv0 + 32 < len) {
        __builtin_prefetch((const char*)K + grow + (uint32_t)(32 * r) * ROWB, 0, 1);
        __builtin_prefetch((const char*)V + grow + (uint32_t)(32 * r) * ROWB, 0, 1);
      }
      asm volatile("s_wait_asynccnt 0" ::: "memory");
    }

    // ---- K B-fragments (shared by both q tiles): row n, dk in table order (8h+s, 16+8h+s-8)
    v16bf kb[2][2];
    #pragma unroll
    for (int n0i = 0; n0i < 2; n0i++)
      #pragma unroll
      for (int c = 0; c < 2; c++) {
        const float* kp = Kf + (n0i * 16 + m) * KVSTR + c * 32 + hh * 8;
        kb[n0i][c] = cvt8x2(kp, kp + 16, 1.0f);
      }

    // ---- per q tile: S = Q K^T, online softmax, stage P to LDS
    #pragma unroll
    for (int qt = 0; qt < 2; qt++) {
      v8f s0 = v8f{0}, s1 = v8f{0};
      s0 = wmma_bf16(qa[qt][0], kb[0][0], s0);
      s0 = wmma_bf16(qa[qt][1], kb[0][1], s0);
      s1 = wmma_bf16(qa[qt][0], kb[1][0], s1);
      s1 = wmma_bf16(qa[qt][1], kb[1][1], s1);

      #pragma unroll
      for (int i = 0; i < 8; i++) {
        float mx   = rmax16(fmaxf(s0[i], s1[i]));
        float mn   = fmaxf(mst[qt][i], mx);
        float corr = fast_exp2(mst[qt][i] - mn);
        float a    = fast_exp2(s0[i] - mn);
        float b    = fast_exp2(s1[i] - mn);
        lst[qt][i] = lst[qt][i] * corr + rsum16(a + b);
        mst[qt][i] = mn;
        of[qt][0][i] *= corr; of[qt][1][i] *= corr;
        of[qt][2][i] *= corr; of[qt][3][i] *= corr;
        const int row = qt * 16 + hh * 8 + i;
        Pb[row * PSTR + m]      = (__bf16)a;
        Pb[row * PSTR + 16 + m] = (__bf16)b;
      }
    }

    // ---- P A-fragments (D-layout -> A-layout via LDS round trip)
    v16bf pa[2];
    #pragma unroll
    for (int qt = 0; qt < 2; qt++) {
      const __bf16* pr = Pb + (qt * 16 + m) * PSTR;
      pa[qt] = cat16(*(const v8bf*)(pr + hh * 8), *(const v8bf*)(pr + 16 + hh * 8));
    }

    // ---- O += P V : V fragment loaded once per d-slab, reused by both q tiles
    #pragma unroll
    for (int t = 0; t < 4; t++) {
      v16bf vb;
      #pragma unroll
      for (int e = 0; e < 16; e++)
        vb[e] = (__bf16)Vf[(hh * 8 + (e & 7) + (e >> 3) * 16) * KVSTR + t * 16 + m];
      of[0][t] = wmma_bf16(pa[0], vb, of[0][t]);
      of[1][t] = wmma_bf16(pa[1], vb, of[1][t]);
    }
  }

  // ---- epilogue: normalize and scatter to dilated output rows
  #pragma unroll
  for (int qt = 0; qt < 2; qt++)
    #pragma unroll
    for (int i = 0; i < 8; i++) {
      const float inv = __builtin_amdgcn_rcpf(lst[qt][i]);
      const int M = qt * 16 + hh * 8 + i;
      const int g = seg0 + off + (q0 + M) * r;
      float* orow = O + ((size_t)g * NHEAD + head) * DHEAD + m;
      orow[0]  = of[qt][0][i] * inv;
      orow[16] = of[qt][1][i] * inv;
      orow[32] = of[qt][2][i] * inv;
      orow[48] = of[qt][3][i] * inv;
    }
}

extern "C" void kernel_launch(void* const* d_in, const int* in_sizes, int n_in,
                              void* d_out, int out_size, void* d_ws, size_t ws_size,
                              hipStream_t stream) {
  (void)in_sizes; (void)n_in; (void)d_ws; (void)ws_size;
  const float* Q = (const float*)d_in[0];
  const float* K = (const float*)d_in[1];
  const float* V = (const float*)d_in[2];
  float* Out = (float*)d_out;

  dilated_zero_kernel<<<(out_size + 255) / 256, 256, 0, stream>>>(Out, out_size);
  // 896 wave-tiles (32 q-rows each), 2 waves per 64-thread block
  dilated_attn_wmma_kernel<<<448, 64, 0, stream>>>(Q, K, V, Out);
}
